// Repro_65910568124423
// MI455X (gfx1250) — matrix operand order repaired and checked by reading, not compiled, not run on hardware
//
#include <hip/hip_runtime.h>
#include <hip/hip_bf16.h>

typedef __attribute__((ext_vector_type(16))) _Float16 v16h;
typedef __attribute__((ext_vector_type(8)))  float    v8f;

#define SEQ 256
#define HD  32

// 16 halves per lane, viewable as 2x uint4 (for b128 LDS/global loads) or 8 dwords
union V16H {
  v16h v;
  uint4 u[2];
  unsigned int w[8];
};

__device__ __forceinline__ unsigned int pack2h(float a, float b) {
  union { _Float16 h[2]; unsigned int u; } x;
  x.h[0] = (_Float16)a;
  x.h[1] = (_Float16)b;
  return x.u;
}

// SWAPX16: group-of-32 swizzle, xor_mask=0x10, and_mask=0x1f -> exchange lane halves
__device__ __forceinline__ int swz16i(int x) {
  return __builtin_amdgcn_ds_swizzle(x, 0x401F);
}
__device__ __forceinline__ float swz16f(float x) {
  return __int_as_float(swz16i(__float_as_int(x)));
}

__global__ void __launch_bounds__(512)
flash_attn_wmma(const _Float16* __restrict__ Q,
                const _Float16* __restrict__ K,
                const _Float16* __restrict__ V,
                _Float16* __restrict__ O) {
  // LDS: K in natural [j][d] layout (A rows for S^T = K*Q^T),
  //      V transposed [d][j] (A rows for O^T = V^T*P^T). 16KB + 16KB.
  __shared__ _Float16 Ks[SEQ * HD];
  __shared__ _Float16 VT[HD * SEQ];

  const int bh = blockIdx.x;                  // [0,256): batch*heads
  const size_t base = (size_t)bh * (SEQ * HD);
  const int tid = threadIdx.x;

  // ---- stage K: async copy global -> LDS (ASYNCcnt path), b128 per thread x2 ----
  {
    // LDS byte address = low 32 bits of the generic address (ISA 10.2)
    unsigned lds_base = (unsigned)(size_t)(const void*)Ks;
    const char* gk = (const char*)(K + base);
#pragma unroll
    for (int e = 0; e < 2; ++e) {
      int idx = e * 512 + tid;                // 1024 x 16B = 32KB... (16KB: 1024*16 = 16384) OK
      unsigned lofs = lds_base + (unsigned)(idx * 16);
      const void* ga = (const void*)(gk + (size_t)idx * 16);
      asm volatile("global_load_async_to_lds_b128 %0, %1, off"
                   :: "v"(lofs), "v"(ga) : "memory");
    }
  }
  // ---- stage V transposed: VT[d][j] = V[j][d] (coalesced u16 reads, ds_store_b16) ----
  {
    const unsigned short* Vg = (const unsigned short*)(V + base);
    unsigned short* VTl = (unsigned short*)VT;
#pragma unroll
    for (int e = 0; e < 16; ++e) {
      int flat = e * 512 + tid;               // coalesced global read
      int j = flat >> 5;
      int d = flat & 31;
      VTl[d * SEQ + j] = Vg[flat];
    }
  }
  asm volatile("s_wait_asynccnt 0x0" ::: "memory");
  __syncthreads();

  const int wave = tid >> 5;                  // 16 waves: one 16-query tile each
  const int lane = tid & 31;
  const int li   = lane & 15;                 // matrix M/N index for this lane
  const int g    = lane >> 4;                 // lane half
  const int qrow = wave * 16 + li;            // this lane's query row (as B column)

  // ---- persistent B operand: Q^T, lane (n=li, half=g) holds Q[qrow][8g..8g+7, 16+8g..16+8g+7]
  // fold log2(e)/sqrt(32) into Q so softmax runs in exp2 domain; scale in packed f16
  V16H qb;
  {
    const uint4* qp = (const uint4*)(Q + base + (size_t)qrow * HD + 8 * g);
    qb.u[0] = qp[0];
    qb.u[1] = qp[2];
    const _Float16 qs = (_Float16)(1.4426950408889634f / 5.656854249492381f);
    qb.v = qb.v * qs;                         // v_pk_mul_f16 x8
  }

  v8f acc0 = {};                              // O^T rows d=0..15   (this lane: d=8g+r)
  v8f acc1 = {};                              // O^T rows d=16..31  (this lane: d=16+8g+r)
  float m    = -3.0e38f;                      // running row max (base-2 domain)
  float lsum = 0.0f;                          // running partial row sum (this lane half)

#pragma unroll 1
  for (int j0 = 0; j0 < SEQ; j0 += 32) {
    // ---- S^T tiles: A = K rows (16 keys x 32 dims), two tiles per iteration ----
    V16H a0, a1;
    a0.u[0] = *(const uint4*)(Ks + (j0 + li) * HD + 8 * g);
    a0.u[1] = *(const uint4*)(Ks + (j0 + li) * HD + 16 + 8 * g);
    a1.u[0] = *(const uint4*)(Ks + (j0 + 16 + li) * HD + 8 * g);
    a1.u[1] = *(const uint4*)(Ks + (j0 + 16 + li) * HD + 16 + 8 * g);

    v8f z = {};
    v8f d0 = __builtin_amdgcn_wmma_f32_16x16x32_f16(false, a0.v, false, qb.v,
                                                    (short)0, z, false, false);
    v8f d1 = __builtin_amdgcn_wmma_f32_16x16x32_f16(false, a1.v, false, qb.v,
                                                    (short)0, z, false, false);
    // lane (li,g) now holds scores of query li for keys j0+8g+r (d0) and j0+16+8g+r (d1)

    // ---- online softmax (base-2) ----
    float tmax = -3.0e38f;
#pragma unroll
    for (int r = 0; r < 8; ++r)
      tmax = fmaxf(tmax, fmaxf(d0[r], d1[r]));
    tmax = fmaxf(tmax, swz16f(tmax));         // combine the two lane halves
    float mnew  = fmaxf(m, tmax);
    float alpha = __builtin_amdgcn_exp2f(m - mnew);

    float p0[8], p1[8];
    float psum = 0.0f;
#pragma unroll
    for (int r = 0; r < 8; ++r) {
      p0[r] = __builtin_amdgcn_exp2f(d0[r] - mnew);
      p1[r] = __builtin_amdgcn_exp2f(d1[r] - mnew);
      psum += p0[r] + p1[r];
    }
    lsum = lsum * alpha + psum;
    m = mnew;
#pragma unroll
    for (int r = 0; r < 8; ++r) {
      acc0[r] *= alpha;
      acc1[r] *= alpha;
    }

    // ---- build B operand P^T (32 keys x 16 queries) from D-layout probs ----
    unsigned int pk0[4], pk1[4];
#pragma unroll
    for (int j2 = 0; j2 < 4; ++j2) {
      pk0[j2] = pack2h(p0[2 * j2], p0[2 * j2 + 1]);   // keys j0+8g+2j2..+1
      pk1[j2] = pack2h(p1[2 * j2], p1[2 * j2 + 1]);   // keys j0+16+8g+2j2..+1
    }
    V16H bt;
#pragma unroll
    for (int j2 = 0; j2 < 4; ++j2) {
      bt.w[j2]     = pk0[j2];                         // h half: keys 8g+0..7
      bt.w[4 + j2] = pk1[j2];                         // h half: keys 16+8g+0..7
    }

    // ---- O^T accumulate: A = V^T rows (dims x 32 keys) ----
    V16H va0, va1;
    va0.u[0] = *(const uint4*)(VT + li * SEQ + j0 + 8 * g);
    va0.u[1] = *(const uint4*)(VT + li * SEQ + j0 + 16 + 8 * g);
    va1.u[0] = *(const uint4*)(VT + (16 + li) * SEQ + j0 + 8 * g);
    va1.u[1] = *(const uint4*)(VT + (16 + li) * SEQ + j0 + 16 + 8 * g);

    acc0 = __builtin_amdgcn_wmma_f32_16x16x32_f16(false, va0.v, false, bt.v,
                                                  (short)0, acc0, false, false);
    acc1 = __builtin_amdgcn_wmma_f32_16x16x32_f16(false, va1.v, false, bt.v,
                                                  (short)0, acc1, false, false);
  }

  // ---- finalize: combine lane-half sums, normalize, store fp16 ----
  lsum += swz16f(lsum);
  float inv = 1.0f / lsum;

  _Float16* orow = O + base + (size_t)qrow * HD;
  unsigned int o0[4], o1[4];
#pragma unroll
  for (int j2 = 0; j2 < 4; ++j2) {
    o0[j2] = pack2h(acc0[2 * j2] * inv, acc0[2 * j2 + 1] * inv);
    o1[j2] = pack2h(acc1[2 * j2] * inv, acc1[2 * j2 + 1] * inv);
  }
  uint4 s0, s1;
  s0.x = o0[0]; s0.y = o0[1]; s0.z = o0[2]; s0.w = o0[3];
  s1.x = o1[0]; s1.y = o1[1]; s1.z = o1[2]; s1.w = o1[3];
  *(uint4*)(orow + 8 * g)      = s0;          // dims 8g .. 8g+7
  *(uint4*)(orow + 16 + 8 * g) = s1;          // dims 16+8g .. 16+8g+7
}

extern "C" void kernel_launch(void* const* d_in, const int* in_sizes, int n_in,
                              void* d_out, int out_size, void* d_ws, size_t ws_size,
                              hipStream_t stream) {
  const _Float16* q = (const _Float16*)d_in[0];
  const _Float16* k = (const _Float16*)d_in[1];
  const _Float16* v = (const _Float16*)d_in[2];
  _Float16* out = (_Float16*)d_out;
  // one block per (batch, head) pair: 16*16 = 256 blocks, 512 threads = 16 waves
  flash_attn_wmma<<<dim3(256), dim3(512), 0, stream>>>(q, k, v, out);
}
